// GNNStackGraph_1958505087002
// MI455X (gfx1250) — matrix operand order repaired and checked by reading, not compiled, not run on hardware
//
#include <hip/hip_runtime.h>
#include <cstdint>
#include <cstddef>

typedef __attribute__((ext_vector_type(16))) _Float16 v16h;
typedef __attribute__((ext_vector_type(8)))  float    v8f;

#define NNODES 50000
#define NEDGES 800000
#define HID 128
#define ODIM 10
#define NGRAPH 512
#define LN_EPS 1e-5f
#define MTILES (NNODES / 16)   // 3125, exact

// ---------------------------------------------------------------------------
// Pack a 128x128 f32 weight (row-major, [K][N]) into the WMMA B-fragment
// layout this file uses with v_wmma_f32_16x16x32_f16 (contiguous k order, same as the A loads):
//   dword index d = ((n0*4 + k0)*32 + lane)*8 + j
//   lane: column n = n0*16 + (lane&15), half-group g = lane>>4
//   halves of dword j: K = k0*32 + g*16 + 2j (+1)
// ---------------------------------------------------------------------------
__global__ void pack_weight_f16(const float* __restrict__ W,
                                uint32_t* __restrict__ out) {
    int d = blockIdx.x * blockDim.x + threadIdx.x;
    if (d >= 8192) return;
    int j    = d & 7;
    int lane = (d >> 3) & 31;
    int k0   = (d >> 8) & 3;
    int n0   = d >> 10;
    int g    = lane >> 4;
    int n    = n0 * 16 + (lane & 15);
    int K0   = k0 * 32 + g * 16 + 2 * j;
    union { _Float16 h[2]; uint32_t u; } p;
    p.h[0] = (_Float16)W[(size_t)K0 * HID + n];
    p.h[1] = (_Float16)W[(size_t)(K0 + 1) * HID + n];
    out[d] = p.u;
}

// ---------------------------------------------------------------------------
// Edge scatter: agg[dst] += cur[src]  (f32 atomics, 4 features per thread)
// ---------------------------------------------------------------------------
__global__ void edge_scatter(const float* __restrict__ cur,
                             const int* __restrict__ src,
                             const int* __restrict__ dst,
                             float* __restrict__ agg) {
    long long gid = (long long)blockIdx.x * blockDim.x + threadIdx.x;
    if (gid >= (long long)NEDGES * 32) return;
    int e = (int)(gid >> 5);
    int c = ((int)gid & 31) * 4;
    int s = src[e], d = dst[e];
    float4 v = *(const float4*)(cur + (size_t)s * HID + c);
    float* ap = agg + (size_t)d * HID + c;
    atomicAdd(ap + 0, v.x);
    atomicAdd(ap + 1, v.y);
    atomicAdd(ap + 2, v.z);
    atomicAdd(ap + 3, v.w);
}

// h = a + b, converted to f16 (GEMM A operand)
__global__ void add_to_f16(const float* __restrict__ a,
                           const float* __restrict__ b,
                           _Float16* __restrict__ out) {
    int gid = blockIdx.x * blockDim.x + threadIdx.x;   // one float4 per thread
    if (gid >= NNODES * HID / 4) return;
    float4 va = ((const float4*)a)[gid];
    float4 vb = ((const float4*)b)[gid];
    union { _Float16 h[4]; uint2 u; } p;
    p.h[0] = (_Float16)(va.x + vb.x);
    p.h[1] = (_Float16)(va.y + vb.y);
    p.h[2] = (_Float16)(va.z + vb.z);
    p.h[3] = (_Float16)(va.w + vb.w);
    ((uint2*)out)[gid] = p.u;
}

// ---------------------------------------------------------------------------
// WMMA GEMM: out[M x 128] = act(A[M x 128] @ Wpacked + bias)
// One wave -> one 16x16 tile; 8 waves/block cover all 128 columns.
// ---------------------------------------------------------------------------
template <bool RELU, typename OUT_T>
__global__ void gemm_wmma_128(const _Float16* __restrict__ A,
                              const uint32_t* __restrict__ Bp,
                              const float* __restrict__ bias,
                              OUT_T* __restrict__ out) {
    const int lane = threadIdx.x & 31;
    const int wv   = threadIdx.x >> 5;        // N-tile 0..7
    const int m0   = blockIdx.x * 16;
    const int g    = lane >> 4;
    const int mr   = lane & 15;
    const int n    = wv * 16 + mr;            // output / bias column

    v8f c = {};
    const _Float16* Arow  = A + (size_t)(m0 + mr) * HID;
    const uint32_t* Bbase = Bp + (size_t)wv * 4 * 32 * 8 + (size_t)lane * 8;

    for (int k0 = 0; k0 < 4; ++k0) {
        union { v16h v; uint32_t u[8]; } a, b;
        const int kb = k0 * 32;
#pragma unroll
        for (int j = 0; j < 4; ++j)
            a.u[j] = *(const uint32_t*)(Arow + kb + g * 16 + 2 * j);
#pragma unroll
        for (int j = 0; j < 4; ++j)
            a.u[4 + j] = *(const uint32_t*)(Arow + kb + g * 16 + 8 + 2 * j);
        const uint32_t* bp = Bbase + (size_t)k0 * 32 * 8;
#pragma unroll
        for (int j = 0; j < 8; ++j) b.u[j] = bp[j];
        c = __builtin_amdgcn_wmma_f32_16x16x32_f16(
                false, a.v, false, b.v, (short)0, c, false, false);
    }

    const float bn = bias[n];
#pragma unroll
    for (int r = 0; r < 8; ++r) {
        int m   = m0 + r + (g ? 8 : 0);
        float v = c[r] + bn;
        if (RELU) v = v > 0.f ? v : 0.f;
        out[(size_t)m * HID + n] = (OUT_T)v;
    }
}

// relu then LayerNorm over 128 features (one row per 128-thread block)
__global__ void relu_ln(const float* __restrict__ h,
                        const float* __restrict__ gam,
                        const float* __restrict__ bet,
                        float* __restrict__ out) {
    __shared__ float red[HID];
    int row = blockIdx.x, tid = threadIdx.x;
    float v = h[(size_t)row * HID + tid];
    v = v > 0.f ? v : 0.f;
    red[tid] = v; __syncthreads();
    for (int s = 64; s > 0; s >>= 1) {
        if (tid < s) red[tid] += red[tid + s];
        __syncthreads();
    }
    float mu = red[0] * (1.0f / HID);
    __syncthreads();
    float d = v - mu;
    red[tid] = d * d; __syncthreads();
    for (int s = 64; s > 0; s >>= 1) {
        if (tid < s) red[tid] += red[tid + s];
        __syncthreads();
    }
    float var = red[0] * (1.0f / HID);
    out[(size_t)row * HID + tid] = d * rsqrtf(var + LN_EPS) * gam[tid] + bet[tid];
}

// sums[batch[i]] += relu(emb[i])
__global__ void pool_scatter(const float* __restrict__ emb,
                             const int* __restrict__ batch,
                             float* __restrict__ sums) {
    int gid = blockIdx.x * blockDim.x + threadIdx.x;
    if (gid >= NNODES * 32) return;
    int i = gid >> 5, c = (gid & 31) * 4;
    int gi = batch[i];
    float4 v = *(const float4*)(emb + (size_t)i * HID + c);
    float* sp = sums + (size_t)gi * HID + c;
    atomicAdd(sp + 0, fmaxf(v.x, 0.f));
    atomicAdd(sp + 1, fmaxf(v.y, 0.f));
    atomicAdd(sp + 2, fmaxf(v.z, 0.f));
    atomicAdd(sp + 3, fmaxf(v.w, 0.f));
}

__global__ void pool_cnt(const int* __restrict__ batch, float* __restrict__ cnt) {
    int i = blockIdx.x * blockDim.x + threadIdx.x;
    if (i < NNODES) atomicAdd(&cnt[batch[i]], 1.0f);
}

__global__ void pool_div(const float* __restrict__ sums,
                         const float* __restrict__ cnt,
                         float* __restrict__ pooled) {
    int gid = blockIdx.x * blockDim.x + threadIdx.x;
    if (gid >= NGRAPH * HID) return;
    int g = gid >> 7;
    pooled[gid] = sums[gid] / fmaxf(cnt[g], 1.0f);
}

// z = pooled @ Wp1 + bp1   (512 x 128, plain f32)
__global__ void post_fc1(const float* __restrict__ pooled,
                         const float* __restrict__ W,
                         const float* __restrict__ b,
                         float* __restrict__ z) {
    __shared__ float r[HID];
    int row = blockIdx.x, tid = threadIdx.x;
    r[tid] = pooled[(size_t)row * HID + tid];
    __syncthreads();
    float acc = b[tid];
    for (int k = 0; k < HID; ++k) acc += r[k] * W[(size_t)k * HID + tid];
    z[(size_t)row * HID + tid] = acc;
}

// logits = z @ Wp2 + bp2, then log_softmax over 10 classes
__global__ void post_fc2_lsm(const float* __restrict__ z,
                             const float* __restrict__ W,
                             const float* __restrict__ b,
                             float* __restrict__ out) {
    __shared__ float r[HID];
    __shared__ float lg[ODIM];
    __shared__ float stat[2];
    int row = blockIdx.x, tid = threadIdx.x;
    r[tid] = z[(size_t)row * HID + tid];
    __syncthreads();
    if (tid < ODIM) {
        float acc = b[tid];
        for (int k = 0; k < HID; ++k) acc += r[k] * W[(size_t)k * ODIM + tid];
        lg[tid] = acc;
    }
    __syncthreads();
    if (tid == 0) {
        float m = lg[0];
        for (int j = 1; j < ODIM; ++j) m = fmaxf(m, lg[j]);
        float s = 0.f;
        for (int j = 0; j < ODIM; ++j) s += __expf(lg[j] - m);
        stat[0] = m; stat[1] = __logf(s);
    }
    __syncthreads();
    if (tid < ODIM) out[(size_t)row * ODIM + tid] = lg[tid] - stat[0] - stat[1];
}

// ---------------------------------------------------------------------------
extern "C" void kernel_launch(void* const* d_in, const int* in_sizes, int n_in,
                              void* d_out, int out_size, void* d_ws, size_t ws_size,
                              hipStream_t stream) {
    const float* x     = (const float*)d_in[0];
    const int*   src   = (const int*)d_in[1];
    const int*   dst   = (const int*)d_in[2];
    const int*   batch = (const int*)d_in[3];
    const float* W1[3] = {(const float*)d_in[4],  (const float*)d_in[8],  (const float*)d_in[12]};
    const float* b1[3] = {(const float*)d_in[5],  (const float*)d_in[9],  (const float*)d_in[13]};
    const float* W2[3] = {(const float*)d_in[6],  (const float*)d_in[10], (const float*)d_in[14]};
    const float* b2[3] = {(const float*)d_in[7],  (const float*)d_in[11], (const float*)d_in[15]};
    const float* lng[2] = {(const float*)d_in[16], (const float*)d_in[18]};
    const float* lnb[2] = {(const float*)d_in[17], (const float*)d_in[19]};
    const float* Wp1 = (const float*)d_in[20];
    const float* bp1 = (const float*)d_in[21];
    const float* Wp2 = (const float*)d_in[22];
    const float* bp2 = (const float*)d_in[23];

    float* emb    = (float*)d_out;                       // 50000*128
    float* logout = (float*)d_out + (size_t)NNODES * HID; // 512*10

    // workspace layout (256B aligned)
    char*  ws  = (char*)d_ws;
    size_t off = 0;
    auto take = [&](size_t bytes) {
        size_t p = off;
        off = (off + bytes + 255) & ~(size_t)255;
        return p;
    };
    const size_t NH = (size_t)NNODES * HID;
    float*     agg  = (float*)(ws + take(NH * 4));
    _Float16*  hin  = (_Float16*)(ws + take(NH * 2));
    _Float16*  t1   = (_Float16*)(ws + take(NH * 2));
    float*     h2   = (float*)(ws + take(NH * 4));
    float*     hcur = (float*)(ws + take(NH * 4));
    uint32_t*  wp1[3]; uint32_t* wp2[3];
    for (int l = 0; l < 3; ++l) {
        wp1[l] = (uint32_t*)(ws + take(8192 * 4));
        wp2[l] = (uint32_t*)(ws + take(8192 * 4));
    }
    float* sums   = (float*)(ws + take((size_t)NGRAPH * HID * 4));
    float* cnt    = (float*)(ws + take((size_t)NGRAPH * 4));
    float* pooled = (float*)(ws + take((size_t)NGRAPH * HID * 4));
    float* z1     = (float*)(ws + take((size_t)NGRAPH * HID * 4));

    // pack all six 128x128 weights into WMMA B layout (f16)
    for (int l = 0; l < 3; ++l) {
        pack_weight_f16<<<8192 / 256, 256, 0, stream>>>(W1[l], wp1[l]);
        pack_weight_f16<<<8192 / 256, 256, 0, stream>>>(W2[l], wp2[l]);
    }

    const float* cur = x;
    const int scat_blocks = (NEDGES * 32 + 255) / 256;
    const int add_blocks  = (int)((NH / 4 + 255) / 256);

    for (int l = 0; l < 3; ++l) {
        hipMemsetAsync(agg, 0, NH * 4, stream);
        edge_scatter<<<scat_blocks, 256, 0, stream>>>(cur, src, dst, agg);
        add_to_f16<<<add_blocks, 256, 0, stream>>>(cur, agg, hin);
        gemm_wmma_128<true, _Float16><<<MTILES, 256, 0, stream>>>(hin, wp1[l], b1[l], t1);
        float* gout = (l == 2) ? emb : h2;
        gemm_wmma_128<false, float><<<MTILES, 256, 0, stream>>>(t1, wp2[l], b2[l], gout);
        if (l < 2) {
            relu_ln<<<NNODES, HID, 0, stream>>>(h2, lng[l], lnb[l], hcur);
            cur = hcur;
        }
    }

    // global mean pool over relu(emb)
    hipMemsetAsync(sums, 0, (size_t)NGRAPH * HID * 4, stream);
    hipMemsetAsync(cnt, 0, (size_t)NGRAPH * 4, stream);
    pool_scatter<<<(NNODES * 32 + 255) / 256, 256, 0, stream>>>(emb, batch, sums);
    pool_cnt<<<(NNODES + 255) / 256, 256, 0, stream>>>(batch, cnt);
    pool_div<<<(NGRAPH * HID + 255) / 256, 256, 0, stream>>>(sums, cnt, pooled);

    // post-MLP + log-softmax
    post_fc1<<<NGRAPH, HID, 0, stream>>>(pooled, Wp1, bp1, z1);
    post_fc2_lsm<<<NGRAPH, HID, 0, stream>>>(z1, Wp2, bp2, logout);
}
